// Decoder_9302899163353
// MI455X (gfx1250) — matrix operand order repaired and checked by reading, not compiled, not run on hardware
//
#include <hip/hip_runtime.h>
#include <hip/hip_bf16.h>

// ---------------------------------------------------------------------------
// Bahdanau-attention GRU decoder for MI455X (gfx1250).
// All GEMMs use v_wmma_f32_16x16x32_f16 (f16 inputs, f32 accumulate).
// proj_key is cached in f16 (33.5 MB) so the per-step attention working set
// (proj_key + encoder_output ~ 167 MB) stays resident in the 192 MB L2.
// ---------------------------------------------------------------------------

typedef __attribute__((ext_vector_type(16))) _Float16 v16h;
typedef __attribute__((ext_vector_type(8)))  float    v8f;

#define B_  64
#define T_  128
#define S_  512
#define E_  256
#define H_  512
#define H2_ 1024

// ---------------------------------------------------------------------------
// Generic WMMA GEMM:  C[M,N] = act( A[M,K] * op(B) + bias )
//   A: f32 row-major (lda), converted to f16 on the fly.
//   B: f32. TRANS_B=0 -> B is K x N row-major (ldb = row stride).
//           TRANS_B=1 -> B is N x K row-major (ldb = row stride); computes A*B^T.
//   ACT: 0 = none, 1 = tanh.  CF16: store result as _Float16 into Ch.
// One wave (32 lanes) owns one 16x16 tile. block = (32,4): 4 N-tiles/block.
// Requires M%16==0, N%64==0, K%32==0, EXEC all ones (no divergence).
// ---------------------------------------------------------------------------
template <bool TRANS_B, bool BIAS, int ACT, bool CF16>
__global__ void gemm_wmma_f16(const float* __restrict__ A, int lda,
                              const float* __restrict__ Bm, int ldb,
                              const float* __restrict__ bias,
                              float* __restrict__ C, _Float16* __restrict__ Ch,
                              int ldc, int M, int N, int K)
{
    const int lane    = threadIdx.x;          // 0..31
    const int halfsel = lane >> 4;            // 0 or 1
    const int sub     = lane & 15;            // row (A) / col (B,C) within tile
    const int tileN   = (blockIdx.x * 4 + threadIdx.y) * 16;
    const int tileM   = blockIdx.y * 16;
    if (tileM >= M || tileN >= N) return;

    v8f acc = {};

    for (int k0 = 0; k0 < K; k0 += 32) {
        // ---- A fragment: 16x32 f16 ----
        // lane<16 : h[0..15] = A[M][k0+0..15]
        // lane>=16: h[0..15] = A[M][k0+16..31]
        v16h a;
        {
            const float* Ap = A + (size_t)(tileM + sub) * lda + k0 + halfsel * 16;
#pragma unroll
            for (int j = 0; j < 8; ++j) a[j]     = (_Float16)Ap[j];
#pragma unroll
            for (int j = 0; j < 8; ++j) a[8 + j] = (_Float16)Ap[8 + j];
        }
        // ---- B fragment: 32x16 f16 ----
        // lane<16 : h[j] = B[k0+j][N],  lane>=16: h[j] = B[k0+16+j][N]
        v16h b;
        if constexpr (TRANS_B) {
            const float* Bp = Bm + (size_t)(tileN + sub) * ldb + k0 + halfsel * 16;
#pragma unroll
            for (int j = 0; j < 16; ++j) b[j] = (_Float16)Bp[j];
        } else {
            const float* Bp = Bm + (size_t)(k0 + halfsel * 16) * ldb + (tileN + sub);
#pragma unroll
            for (int j = 0; j < 16; ++j) b[j] = (_Float16)Bp[(size_t)j * ldb];
        }
        acc = __builtin_amdgcn_wmma_f32_16x16x32_f16(
            /*neg_a=*/false, a, /*neg_b=*/false, b,
            /*c_mod=*/(short)0, acc, /*reuse_a=*/false, /*reuse_b=*/false);
    }

    // ---- epilogue: lane<16 -> rows 0..7, lane>=16 -> rows 8..15 ----
    const int col   = tileN + sub;
    const int rbase = tileM + halfsel * 8;
#pragma unroll
    for (int i = 0; i < 8; ++i) {
        float v = acc[i];
        if constexpr (BIAS) v += bias[col];
        if constexpr (ACT == 1) v = tanhf(v);
        const size_t idx = (size_t)(rbase + i) * ldc + col;
        if constexpr (CF16) Ch[idx] = (_Float16)v;
        else                C[idx]  = v;
    }
}

// ---------------------------------------------------------------------------
// Attention for one timestep: scores = v . tanh(q + proj_key), softmax over S,
// context = attn @ encoder_output.  One block per batch row, 256 threads.
// Also stages x = [emb|ctx] and pre_in = [emb|h|ctx] (emb/ctx parts).
// ---------------------------------------------------------------------------
__global__ void attn_kernel(const _Float16* __restrict__ pk,  // [B,S,H] f16
                            const float* __restrict__ q,      // [B,H]
                            const float* __restrict__ v_en,   // [H]
                            const float* __restrict__ enc,    // [B,S,2H]
                            const float* __restrict__ emb,    // [B,T,E]
                            int t,
                            float* __restrict__ att_probs,    // [B,T,S]
                            float* __restrict__ x_buf,        // [B,E+2H]
                            float* __restrict__ pre_buf)      // [B,E+H+2H]
{
    const int b = blockIdx.x;
    const int tid = threadIdx.x;            // 0..255
    const int wave = tid >> 5, lane = tid & 31;

    __shared__ float sq[H_];
    __shared__ float sc[S_];
    __shared__ float sred[8];
    __shared__ float sred2[8];

    for (int i = tid; i < H_; i += 256) sq[i] = q[b * H_ + i];
    __syncthreads();

    // ---- energies ----
    for (int s = wave; s < S_; s += 8) {
        const _Float16* p = pk + ((size_t)b * S_ + s) * H_;
        float acc = 0.f;
        for (int h = lane; h < H_; h += 32)
            acc += v_en[h] * tanhf(sq[h] + (float)p[h]);
#pragma unroll
        for (int off = 16; off > 0; off >>= 1) acc += __shfl_xor(acc, off, 32);
        if (lane == 0) sc[s] = acc;
    }
    __syncthreads();

    // ---- softmax max ----
    float m = -1e30f;
    for (int i = tid; i < S_; i += 256) m = fmaxf(m, sc[i]);
#pragma unroll
    for (int off = 16; off > 0; off >>= 1) m = fmaxf(m, __shfl_xor(m, off, 32));
    if (lane == 0) sred[wave] = m;
    __syncthreads();
    if (tid == 0) {
        float mm = sred[0];
#pragma unroll
        for (int i = 1; i < 8; ++i) mm = fmaxf(mm, sred[i]);
        sred[0] = mm;
    }
    __syncthreads();
    m = sred[0];

    // ---- exp + sum ----
    float ssum = 0.f;
    for (int i = tid; i < S_; i += 256) {
        float e = __expf(sc[i] - m);
        sc[i] = e;
        ssum += e;
    }
#pragma unroll
    for (int off = 16; off > 0; off >>= 1) ssum += __shfl_xor(ssum, off, 32);
    if (lane == 0) sred2[wave] = ssum;
    __syncthreads();
    if (tid == 0) {
        float tt = 0.f;
#pragma unroll
        for (int i = 0; i < 8; ++i) tt += sred2[i];
        sred2[0] = tt;
    }
    __syncthreads();
    const float inv = 1.f / sred2[0];

    for (int i = tid; i < S_; i += 256) {
        float a = sc[i] * inv;
        sc[i] = a;
        att_probs[((size_t)b * T_ + t) * S_ + i] = a;
    }
    __syncthreads();

    // ---- context = attn @ enc  (2H columns, coalesced across threads) ----
    for (int c = tid; c < H2_; c += 256) {
        const float* ep = enc + (size_t)b * S_ * H2_ + c;
        float acc = 0.f;
        for (int s = 0; s < S_; ++s) acc += sc[s] * ep[(size_t)s * H2_];
        x_buf[b * (E_ + H2_) + E_ + c]        = acc;
        pre_buf[b * (E_ + H_ + H2_) + E_ + H_ + c] = acc;
    }
    // ---- stage embedding slice ----
    const float* e = emb + ((size_t)b * T_ + t) * E_;
    for (int i = tid; i < E_; i += 256) {
        x_buf[b * (E_ + H2_) + i]        = e[i];
        pre_buf[b * (E_ + H_ + H2_) + i] = e[i];
    }
}

// ---------------------------------------------------------------------------
// Fused GRU gate elementwise (PyTorch gate order r,z,n). One thread per (b,h).
// ---------------------------------------------------------------------------
__global__ void gru_kernel(const float* __restrict__ gi,   // [B,3H]
                           const float* __restrict__ gh,   // [B,3H]
                           const float* __restrict__ b_ih, // [3H]
                           const float* __restrict__ b_hh, // [3H]
                           float* __restrict__ h,          // [B,H] in/out
                           float* __restrict__ pre_buf,    // [B,E+H+2H]
                           float* __restrict__ dec_states, // [B,T,H]
                           float* __restrict__ hidden_out, // [B,H]
                           int t)
{
    const int idx = blockIdx.x * blockDim.x + threadIdx.x;   // B*H
    const int b = idx >> 9, hh = idx & (H_ - 1);
    const int g = b * 3 * H_;

    const float gir = gi[g + hh]          + b_ih[hh];
    const float giz = gi[g + H_ + hh]     + b_ih[H_ + hh];
    const float gin = gi[g + 2 * H_ + hh] + b_ih[2 * H_ + hh];
    const float ghr = gh[g + hh]          + b_hh[hh];
    const float ghz = gh[g + H_ + hh]     + b_hh[H_ + hh];
    const float ghn = gh[g + 2 * H_ + hh] + b_hh[2 * H_ + hh];

    const float r = 1.f / (1.f + __expf(-(gir + ghr)));
    const float z = 1.f / (1.f + __expf(-(giz + ghz)));
    const float n = tanhf(gin + r * ghn);
    const float hn = (1.f - z) * n + z * h[idx];

    h[idx] = hn;
    pre_buf[b * (E_ + H_ + H2_) + E_ + hh] = hn;
    dec_states[((size_t)b * T_ + t) * H_ + hh] = hn;
    if (t == T_ - 1) hidden_out[idx] = hn;
}

// ---------------------------------------------------------------------------
extern "C" void kernel_launch(void* const* d_in, const int* in_sizes, int n_in,
                              void* d_out, int out_size, void* d_ws, size_t ws_size,
                              hipStream_t stream)
{
    const float* trg_embed  = (const float*)d_in[0];   // [B,T,E]
    const float* enc_out    = (const float*)d_in[1];   // [B,S,2H]
    const float* enc_hidden = (const float*)d_in[2];   // [B,2H]
    const float* Wk         = (const float*)d_in[3];   // [2H,H]
    const float* Wq         = (const float*)d_in[4];   // [H,H]
    const float* v_energy   = (const float*)d_in[5];   // [H]
    const float* W_bridge   = (const float*)d_in[6];   // [2H,H]
    const float* b_bridge   = (const float*)d_in[7];   // [H]
    const float* W_ih       = (const float*)d_in[8];   // [3H,E+2H]
    const float* W_hh       = (const float*)d_in[9];   // [3H,H]
    const float* b_ih       = (const float*)d_in[10];  // [3H]
    const float* b_hh       = (const float*)d_in[11];  // [3H]
    const float* W_pre      = (const float*)d_in[12];  // [E+3H,H]

    float* out        = (float*)d_out;
    float* dec_states = out;                                        // [B,T,H]
    float* hidden_out = dec_states + (size_t)B_ * T_ * H_;          // [1,B,H]
    float* att_vec    = hidden_out + (size_t)B_ * H_;               // [B,T,H]
    float* att_probs  = att_vec    + (size_t)B_ * T_ * H_;          // [B,T,S]

    // ---- workspace carve-out (≈35.4 MB) ----
    char* w = (char*)d_ws;
    auto take = [&](size_t bytes) { char* p = w; w += (bytes + 255) & ~(size_t)255; return p; };
    _Float16* pk  = (_Float16*)take((size_t)B_ * S_ * H_ * sizeof(_Float16)); // proj_key f16
    float* hbuf   = (float*)take((size_t)B_ * H_ * sizeof(float));
    float* qbuf   = (float*)take((size_t)B_ * H_ * sizeof(float));
    float* xbuf   = (float*)take((size_t)B_ * (E_ + H2_) * sizeof(float));
    float* prebuf = (float*)take((size_t)B_ * (E_ + H_ + H2_) * sizeof(float));
    float* gibuf  = (float*)take((size_t)B_ * 3 * H_ * sizeof(float));
    float* ghbuf  = (float*)take((size_t)B_ * 3 * H_ * sizeof(float));

    const dim3 blk(32, 4, 1);   // 4 wave32s, one 16x16 WMMA tile each

    // h0 = tanh(enc_hidden @ W_bridge + b_bridge)        [64x512], K=1024
    gemm_wmma_f16<false, true, 1, false><<<dim3(H_ / 64, B_ / 16), blk, 0, stream>>>(
        enc_hidden, H2_, W_bridge, H_, b_bridge, hbuf, nullptr, H_, B_, H_, H2_);

    // proj_key = enc_out @ Wk -> f16                     [32768x512], K=1024
    gemm_wmma_f16<false, false, 0, true><<<dim3(H_ / 64, (B_ * S_) / 16), blk, 0, stream>>>(
        enc_out, H2_, Wk, H_, nullptr, nullptr, pk, H_, B_ * S_, H_, H2_);

    for (int t = 0; t < T_; ++t) {
        // q = h @ Wq                                     [64x512], K=512
        gemm_wmma_f16<false, false, 0, false><<<dim3(H_ / 64, B_ / 16), blk, 0, stream>>>(
            hbuf, H_, Wq, H_, nullptr, qbuf, nullptr, H_, B_, H_, H_);

        // attention + softmax + context + staging
        attn_kernel<<<B_, 256, 0, stream>>>(pk, qbuf, v_energy, enc_out, trg_embed,
                                            t, att_probs, xbuf, prebuf);

        // gi = x @ W_ih^T                                [64x1536], K=1280
        gemm_wmma_f16<true, false, 0, false><<<dim3((3 * H_) / 64, B_ / 16), blk, 0, stream>>>(
            xbuf, E_ + H2_, W_ih, E_ + H2_, nullptr, gibuf, nullptr, 3 * H_, B_, 3 * H_, E_ + H2_);

        // gh = h @ W_hh^T                                [64x1536], K=512
        gemm_wmma_f16<true, false, 0, false><<<dim3((3 * H_) / 64, B_ / 16), blk, 0, stream>>>(
            hbuf, H_, W_hh, H_, nullptr, ghbuf, nullptr, 3 * H_, B_, 3 * H_, H_);

        // fused GRU gates -> h_new (also writes dec_states, hidden at t=T-1)
        gru_kernel<<<(B_ * H_) / 256, 256, 0, stream>>>(gibuf, ghbuf, b_ih, b_hh,
                                                        hbuf, prebuf, dec_states,
                                                        hidden_out, t);

        // att_vec[:,t,:] = tanh(pre_in @ W_pre)          [64x512], K=1792
        gemm_wmma_f16<false, false, 1, false><<<dim3(H_ / 64, B_ / 16), blk, 0, stream>>>(
            prebuf, E_ + H_ + H2_, W_pre, H_, nullptr, att_vec + (size_t)t * H_,
            nullptr, T_ * H_, B_, H_, E_ + H_ + H2_);
    }
}
